// VideoGuidedTextRefiner_19310172963217
// MI455X (gfx1250) — matrix operand order repaired and checked by reading, not compiled, not run on hardware
//
#include <hip/hip_runtime.h>
#include <hip/hip_bf16.h>
#include <type_traits>

typedef __attribute__((ext_vector_type(16))) _Float16 v16h;
typedef __attribute__((ext_vector_type(8)))  _Float16 v8h;
typedef __attribute__((ext_vector_type(4)))  _Float16 v4h;
typedef __attribute__((ext_vector_type(8)))  float    v8f;
typedef __attribute__((ext_vector_type(4)))  float    v4f;

#define D_DIM 1024
#define NQ    4096
#define NV    1024
#define NHEAD 16
#define DH    64

// ---------------------------------------------------------------------------
// Tiled GEMM:  out[M,1024] = A[M,1024] @ W[1024,1024] + bias
// BM=128, BN=64, BK=32. 256 threads = 8 waves; wave -> 32x32 tile (2x2 WMMA).
// f32 A:  register double-buffered staging (conversion fused).
// f16 A:  async global->LDS DMA into a ping-pong buffer (no VGPR roundtrip),
//         synchronized with s_wait_asynccnt + workgroup barrier.
// ---------------------------------------------------------------------------
template <typename AT, bool OUT_F32>
__global__ __launch_bounds__(256) void gemm_bias_kernel(
    const AT* __restrict__ A, const float* __restrict__ W,
    const float* __restrict__ bias, void* __restrict__ outp)
{
    constexpr int BM = 128, BN = 64, BK = 32, LDA = 40, LDB = 40;
    constexpr bool ASYNC_A = !std::is_same_v<AT, float>;
    constexpr int NBUF = ASYNC_A ? 2 : 1;
    __shared__ alignas(16) _Float16 As[NBUF][BM][LDA];
    __shared__ alignas(16) _Float16 Bs[BN][LDB];   // Bs[n][k]

    const int tid  = threadIdx.x;
    const int lane = tid & 31;
    const int wave = tid >> 5;
    const int m0   = blockIdx.x * BM;
    const int n0   = blockIdx.y * BN;
    const int wm   = (wave & 3) * 32;
    const int wn   = (wave >> 2) * 32;
    const int lr   = lane & 15;
    const int hi   = lane >> 4;
    const int klo  = hi * 8;
    const int kb   = hi * 16;

    v4f aregf[4];          // f32-A staging regs
    v4f breg[2];           // W staging regs

    auto loadAreg = [&](int k0) {
        #pragma unroll
        for (int i = 0; i < 4; ++i) {
            const int idx = tid + i * 256;          // 1024 float4 chunks
            const int m = idx >> 3, kq = (idx & 7) * 4;
            aregf[i] = *(const v4f*)((const float*)A + (size_t)(m0 + m) * D_DIM + k0 + kq);
        }
    };
    auto storeAreg = [&]() {
        #pragma unroll
        for (int i = 0; i < 4; ++i) {
            const int idx = tid + i * 256;
            const int m = idx >> 3, kq = (idx & 7) * 4;
            v4h hv;
            #pragma unroll
            for (int j = 0; j < 4; ++j) hv[j] = (_Float16)aregf[i][j];
            *(v4h*)&As[0][m][kq] = hv;
        }
    };
    auto asyncA = [&](int buf, int k0) {
        #pragma unroll
        for (int i = 0; i < 2; ++i) {
            const int idx = tid + i * 256;          // 512 b128 chunks
            const int m = idx >> 2, k8 = (idx & 3) * 8;
            const AT* gp = A + (size_t)(m0 + m) * D_DIM + k0 + k8;
            const unsigned lds = (unsigned)(uintptr_t)&As[buf][m][k8];
            asm volatile("global_load_async_to_lds_b128 %0, %1, off"
                         :: "v"(lds), "v"(gp) : "memory");
        }
    };
    auto loadB = [&](int k0) {
        #pragma unroll
        for (int i = 0; i < 2; ++i) {
            const int idx = tid + i * 256;          // 512 float4 chunks
            const int k = idx >> 4, n4 = (idx & 15) * 4;
            breg[i] = *(const v4f*)(W + (size_t)(k0 + k) * D_DIM + n0 + n4);
        }
    };
    auto storeB = [&]() {
        #pragma unroll
        for (int i = 0; i < 2; ++i) {
            const int idx = tid + i * 256;
            const int k = idx >> 4, n4 = (idx & 15) * 4;
            #pragma unroll
            for (int j = 0; j < 4; ++j) Bs[n4 + j][k] = (_Float16)breg[i][j];
        }
    };

    v8f c[2][2] = {};
    if constexpr (ASYNC_A) asyncA(0, 0); else loadAreg(0);
    loadB(0);

    for (int k0 = 0, it = 0; k0 < D_DIM; k0 += BK, ++it) {
        const int buf = ASYNC_A ? (it & 1) : 0;
        if constexpr (!ASYNC_A) storeAreg();
        storeB();
        if constexpr (ASYNC_A)
            asm volatile("s_wait_asynccnt 0" ::: "memory");
        __syncthreads();
        if (k0 + BK < D_DIM) {
            if constexpr (ASYNC_A) asyncA(buf ^ 1, k0 + BK); else loadAreg(k0 + BK);
            loadB(k0 + BK);
        }

        v16h a[2], b[2];
        #pragma unroll
        for (int s = 0; s < 2; ++s) {
            const _Float16* ap = &As[buf][wm + s * 16 + lr][0];
            v8h alo = *(const v8h*)(ap + klo);
            v8h ahi = *(const v8h*)(ap + 16 + klo);
            #pragma unroll
            for (int e = 0; e < 8; ++e) { a[s][e] = alo[e]; a[s][8 + e] = ahi[e]; }
        }
        #pragma unroll
        for (int t = 0; t < 2; ++t) {
            const _Float16* bp = &Bs[wn + t * 16 + lr][klo];
            v8h blo = *(const v8h*)(bp);
            v8h bhi = *(const v8h*)(bp + 16);
            #pragma unroll
            for (int e = 0; e < 8; ++e) { b[t][e] = blo[e]; b[t][8 + e] = bhi[e]; }
        }
        #pragma unroll
        for (int s = 0; s < 2; ++s)
            #pragma unroll
            for (int t = 0; t < 2; ++t)
                c[s][t] = __builtin_amdgcn_wmma_f32_16x16x32_f16(
                    false, a[s], false, b[t], (short)0, c[s][t], false, false);
        __syncthreads();
    }

    #pragma unroll
    for (int s = 0; s < 2; ++s)
        #pragma unroll
        for (int t = 0; t < 2; ++t) {
            const int col = n0 + wn + t * 16 + lr;
            const float bv = bias[col];
            #pragma unroll
            for (int r = 0; r < 8; ++r) {
                const int row = m0 + wm + s * 16 + r + hi * 8;
                const float val = c[s][t][r] + bv;
                if constexpr (OUT_F32)
                    ((float*)outp)[(size_t)row * D_DIM + col] = val;
                else
                    ((_Float16*)outp)[(size_t)row * D_DIM + col] = (_Float16)val;
            }
        }
}

// ---------------------------------------------------------------------------
// Flash attention: block = (head, 64 q-rows); 4 waves, each a 16-row q tile.
// K tiles: async global->LDS DMA, ping-pong buffered. V tiles: register
// double-buffered (transposed on LDS store). Online softmax with unrolled
// register state; row-sum via WMMA against an all-ones B fragment.
// ---------------------------------------------------------------------------
__global__ __launch_bounds__(128) void attn_kernel(
    const _Float16* __restrict__ Qh, const _Float16* __restrict__ Kh,
    const _Float16* __restrict__ Vh, _Float16* __restrict__ Ctx)
{
    constexpr int LDK = 72, LDV = 40, LDP = 40;
    __shared__ alignas(16) _Float16 Ks[2][32][LDK];  // ping-pong [kv_local][dh]
    __shared__ alignas(16) _Float16 Vt[64][LDV];     // [dh][kv_local]
    __shared__ alignas(16) _Float16 Pw[4][16][LDP];  // per-wave P scratch

    const int tid   = threadIdx.x;
    const int lane  = tid & 31;
    const int wave  = tid >> 5;
    const int h     = blockIdx.y;
    const int qbase = blockIdx.x * 64 + wave * 16;
    const int lr    = lane & 15;
    const int hi    = lane >> 4;
    const int klo   = hi * 8;
    const int kb16  = hi * 16;

    // Q fragments with 1/sqrt(DH) folded in
    v16h qf[2];
    #pragma unroll
    for (int kc = 0; kc < 2; ++kc) {
        const _Float16* qp = Qh + (size_t)(qbase + lr) * D_DIM + h * DH + kc * 32;
        v8h lo  = *(const v8h*)(qp + klo);
        v8h hi8 = *(const v8h*)(qp + 16 + klo);
        #pragma unroll
        for (int e = 0; e < 8; ++e) {
            qf[kc][e]     = lo[e]  * (_Float16)0.125f;
            qf[kc][8 + e] = hi8[e] * (_Float16)0.125f;
        }
    }

    v16h ones;
    #pragma unroll
    for (int e = 0; e < 16; ++e) ones[e] = (_Float16)1.0f;

    float mrow[8], lrow[8];
    #pragma unroll
    for (int r = 0; r < 8; ++r) { mrow[r] = -1e30f; lrow[r] = 0.f; }
    v8f acc[4] = {};

    v8h vreg[2];
    auto asyncK = [&](int buf, int kv) {
        #pragma unroll
        for (int i = 0; i < 2; ++i) {
            const int chunk = tid + i * 128;            // 256 b128 chunks
            const int r = chunk >> 3, c8 = (chunk & 7) * 8;
            const _Float16* gp = Kh + (size_t)(kv + r) * D_DIM + h * DH + c8;
            const unsigned lds = (unsigned)(uintptr_t)&Ks[buf][r][c8];
            asm volatile("global_load_async_to_lds_b128 %0, %1, off"
                         :: "v"(lds), "v"(gp) : "memory");
        }
    };
    auto loadV = [&](int kv) {
        #pragma unroll
        for (int i = 0; i < 2; ++i) {
            const int chunk = tid + i * 128;
            const int r = chunk >> 3, c8 = (chunk & 7) * 8;
            const size_t off = (size_t)(kv + r) * D_DIM + h * DH + c8;
            vreg[i] = *(const v8h*)(Vh + off);
            if (kv + 32 < NV)
                __builtin_prefetch((const void*)(Vh + off + (size_t)32 * D_DIM), 0, 1);
        }
    };
    auto storeV = [&]() {
        #pragma unroll
        for (int i = 0; i < 2; ++i) {
            const int chunk = tid + i * 128;
            const int r = chunk >> 3, c8 = (chunk & 7) * 8;
            #pragma unroll
            for (int j = 0; j < 8; ++j) Vt[c8 + j][r] = vreg[i][j];
        }
    };

    asyncK(0, 0);
    loadV(0);
    for (int kv = 0, it = 0; kv < NV; kv += 32, ++it) {
        const int buf = it & 1;
        storeV();
        asm volatile("s_wait_asynccnt 0" ::: "memory");  // my K DMA landed
        __syncthreads();                                 // everyone's landed
        if (kv + 32 < NV) { asyncK(buf ^ 1, kv + 32); loadV(kv + 32); }

        // scores S (16x32) = (Q/8) (16x64) @ K^T (64x32)
        v8f s[2] = {};
        #pragma unroll
        for (int kc = 0; kc < 2; ++kc)
            #pragma unroll
            for (int t = 0; t < 2; ++t) {
                const _Float16* bp = &Ks[buf][t * 16 + lr][kc * 32 + klo];
                v8h blo = *(const v8h*)bp;
                v8h bhi = *(const v8h*)(bp + 16);
                v16h bb;
                #pragma unroll
                for (int e = 0; e < 8; ++e) { bb[e] = blo[e]; bb[8 + e] = bhi[e]; }
                s[t] = __builtin_amdgcn_wmma_f32_16x16x32_f16(
                    false, qf[kc], false, bb, (short)0, s[t], false, false);
            }

        // online softmax: max via lane shuffles, sum via WMMA below
        float alpha[8];
        #pragma unroll
        for (int r = 0; r < 8; ++r) {
            const float v0 = s[0][r], v1 = s[1][r];
            float mx = fmaxf(v0, v1);
            #pragma unroll
            for (int off = 8; off; off >>= 1) mx = fmaxf(mx, __shfl_xor(mx, off, 32));
            const float mn = fmaxf(mrow[r], mx);
            alpha[r] = __expf(mrow[r] - mn);
            mrow[r]  = mn;
            const float p0 = __expf(v0 - mn);
            const float p1 = __expf(v1 - mn);
            Pw[wave][r + hi * 8][lr]      = (_Float16)p0;
            Pw[wave][r + hi * 8][16 + lr] = (_Float16)p1;
            #pragma unroll
            for (int f = 0; f < 4; ++f) acc[f][r] *= alpha[r];
        }
        asm volatile("s_wait_dscnt 0" ::: "memory");

        // re-read P as an A-fragment (16x32)
        v16h pf;
        {
            const _Float16* pp = &Pw[wave][lr][0];
            v8h lo  = *(const v8h*)(pp + klo);
            v8h hi8 = *(const v8h*)(pp + 16 + klo);
            #pragma unroll
            for (int e = 0; e < 8; ++e) { pf[e] = lo[e]; pf[8 + e] = hi8[e]; }
        }

        // row-sum of P via WMMA against ones (broadcast into every lane)
        v8f rs = {};
        rs = __builtin_amdgcn_wmma_f32_16x16x32_f16(
            false, pf, false, ones, (short)0, rs, false, false);
        #pragma unroll
        for (int r = 0; r < 8; ++r) lrow[r] = lrow[r] * alpha[r] + rs[r];

        // ctx += P (16x32) @ V (32x64)
        #pragma unroll
        for (int f = 0; f < 4; ++f) {
            const _Float16* bp = &Vt[f * 16 + lr][klo];
            v8h blo = *(const v8h*)bp;
            v8h bhi = *(const v8h*)(bp + 16);
            v16h bb;
            #pragma unroll
            for (int e = 0; e < 8; ++e) { bb[e] = blo[e]; bb[8 + e] = bhi[e]; }
            acc[f] = __builtin_amdgcn_wmma_f32_16x16x32_f16(
                false, pf, false, bb, (short)0, acc[f], false, false);
        }
        __syncthreads();
    }

    #pragma unroll
    for (int f = 0; f < 4; ++f) {
        const int col = h * DH + f * 16 + lr;
        #pragma unroll
        for (int r = 0; r < 8; ++r) {
            const int row = qbase + r + hi * 8;
            Ctx[(size_t)row * D_DIM + col] = (_Float16)(acc[f][r] / lrow[r]);
        }
    }
}

// ---------------------------------------------------------------------------
// Fused residual + LN1 + LN2, f16 output for the final GEMM.
// ---------------------------------------------------------------------------
__device__ __forceinline__ float block_sum(float v, float* red, int tid)
{
    #pragma unroll
    for (int off = 16; off; off >>= 1) v += __shfl_xor(v, off, 32);
    if ((tid & 31) == 0) red[tid >> 5] = v;
    __syncthreads();
    if (tid < 32) {
        float t = (tid < 8) ? red[tid] : 0.f;
        #pragma unroll
        for (int off = 4; off; off >>= 1) t += __shfl_xor(t, off, 32);
        if (tid == 0) red[0] = t;
    }
    __syncthreads();
    float r = red[0];
    __syncthreads();
    return r;
}

__global__ __launch_bounds__(256) void ln2x_kernel(
    const float* __restrict__ att, const float* __restrict__ resid,
    const float* __restrict__ g1, const float* __restrict__ b1,
    const float* __restrict__ g2, const float* __restrict__ b2,
    _Float16* __restrict__ out)
{
    __shared__ float red[8];
    const int row = blockIdx.x;
    const int tid = threadIdx.x;

    float x[4];
    float s = 0.f;
    #pragma unroll
    for (int i = 0; i < 4; ++i) {
        const int c = tid + i * 256;
        x[i] = att[(size_t)row * D_DIM + c] + resid[(size_t)row * D_DIM + c];
        s += x[i];
    }
    const float mu1 = block_sum(s, red, tid) * (1.f / D_DIM);
    float vs = 0.f;
    #pragma unroll
    for (int i = 0; i < 4; ++i) { const float d = x[i] - mu1; vs += d * d; }
    const float r1 = rsqrtf(block_sum(vs, red, tid) * (1.f / D_DIM) + 1e-12f);

    float y[4];
    float s2 = 0.f;
    #pragma unroll
    for (int i = 0; i < 4; ++i) {
        const int c = tid + i * 256;
        y[i] = (x[i] - mu1) * r1 * g1[c] + b1[c];
        s2 += y[i];
    }
    const float mu2 = block_sum(s2, red, tid) * (1.f / D_DIM);
    float vs2 = 0.f;
    #pragma unroll
    for (int i = 0; i < 4; ++i) { const float d = y[i] - mu2; vs2 += d * d; }
    const float r2 = rsqrtf(block_sum(vs2, red, tid) * (1.f / D_DIM) + 1e-5f);

    #pragma unroll
    for (int i = 0; i < 4; ++i) {
        const int c = tid + i * 256;
        out[(size_t)row * D_DIM + c] = (_Float16)((y[i] - mu2) * r2 * g2[c] + b2[c]);
    }
}

// ---------------------------------------------------------------------------
extern "C" void kernel_launch(void* const* d_in, const int* in_sizes, int n_in,
                              void* d_out, int out_size, void* d_ws, size_t ws_size,
                              hipStream_t stream)
{
    const float* query = (const float*)d_in[0];
    const float* video = (const float*)d_in[1];
    const float* Wq = (const float*)d_in[2];  const float* bq = (const float*)d_in[3];
    const float* Wk = (const float*)d_in[4];  const float* bk = (const float*)d_in[5];
    const float* Wv = (const float*)d_in[6];  const float* bv = (const float*)d_in[7];
    const float* Wo = (const float*)d_in[8];  const float* bo = (const float*)d_in[9];
    const float* g1 = (const float*)d_in[10]; const float* b1 = (const float*)d_in[11];
    const float* g2 = (const float*)d_in[12]; const float* b2 = (const float*)d_in[13];
    const float* Wl = (const float*)d_in[14]; const float* bl = (const float*)d_in[15];

    char* ws = (char*)d_ws;
    _Float16* Qh  = (_Float16*)(ws);                                   // 8 MB
    _Float16* Kh  = (_Float16*)(ws + (size_t)NQ * D_DIM * 2);          // 2 MB
    _Float16* Vh  = (_Float16*)(ws + (size_t)(NQ + NV) * D_DIM * 2);   // 2 MB
    _Float16* Ctx = (_Float16*)(ws + (size_t)(NQ + 2 * NV) * D_DIM * 2);
    float*    Att = (float*)(ws + (size_t)(2 * NQ + 2 * NV) * D_DIM * 2);
    _Float16* X2  = (_Float16*)(ws + (size_t)(2 * NQ + 2 * NV) * D_DIM * 2
                                   + (size_t)NQ * D_DIM * 4);

    const dim3 blk(256);
    gemm_bias_kernel<float, false><<<dim3(NQ / 128, 16), blk, 0, stream>>>(query, Wq, bq, Qh);
    gemm_bias_kernel<float, false><<<dim3(NV / 128, 16), blk, 0, stream>>>(video, Wk, bk, Kh);
    gemm_bias_kernel<float, false><<<dim3(NV / 128, 16), blk, 0, stream>>>(video, Wv, bv, Vh);
    attn_kernel<<<dim3(NQ / 64, NHEAD), dim3(128), 0, stream>>>(Qh, Kh, Vh, Ctx);
    gemm_bias_kernel<_Float16, true><<<dim3(NQ / 128, 16), blk, 0, stream>>>(Ctx, Wo, bo, Att);
    ln2x_kernel<<<dim3(NQ), blk, 0, stream>>>(Att, query, g1, b1, g2, b2, X2);
    gemm_bias_kernel<_Float16, true><<<dim3(NQ / 128, 16), blk, 0, stream>>>(X2, Wl, bl, (float*)d_out);
}
